// LSTMModel_34016140985079
// MI455X (gfx1250) — matrix operand order repaired and checked by reading, not compiled, not run on hardware
//
#include <hip/hip_runtime.h>
#include <hip/hip_fp16.h>

typedef __attribute__((ext_vector_type(16))) _Float16 v16h;
typedef __attribute__((ext_vector_type(8)))  _Float16 v8h;
typedef __attribute__((ext_vector_type(8)))  float    v8f;
typedef __attribute__((ext_vector_type(4)))  unsigned int v4u;
typedef __attribute__((ext_vector_type(8)))  int      v8i;
typedef __attribute__((ext_vector_type(4)))  int      v4i;

#define HID   128   // H
#define INP   256   // I
#define TT    512   // T
#define BB    256   // batch
#define G4    512   // 4*H

#if defined(__gfx1250__) && __has_builtin(__builtin_amdgcn_tensor_load_to_lds)
#define USE_TDM 1
#else
#define USE_TDM 0
#endif

// ---------------------------------------------------------------------------
// helpers
// ---------------------------------------------------------------------------
__device__ __forceinline__ float sigmoid_f(float x) {
  return 1.0f / (1.0f + __expf(-x));
}
__device__ __forceinline__ float tanh_f(float x) {
  // stable: tanh(x) = sign(x) * (1 - 2/(e^{2|x|}+1));  e->inf gives 1 cleanly
  float ax = fabsf(x);
  float e  = __expf(2.0f * ax);
  float t  = 1.0f - 2.0f / (e + 1.0f);
  return copysignf(t, x);
}

// ---------------------------------------------------------------------------
// kernel 0a: f32 -> f16 conversion (8 elems / thread, sizes are multiples of 8)
// ---------------------------------------------------------------------------
__global__ void cvt_f32_to_f16(const float* __restrict__ src,
                               _Float16* __restrict__ dst, int n) {
  int i = (blockIdx.x * blockDim.x + threadIdx.x) * 8;
  if (i + 8 <= n) {
    float4 a = *(const float4*)(src + i);
    float4 b = *(const float4*)(src + i + 4);
    v8h o;
    o[0] = (_Float16)a.x; o[1] = (_Float16)a.y;
    o[2] = (_Float16)a.z; o[3] = (_Float16)a.w;
    o[4] = (_Float16)b.x; o[5] = (_Float16)b.y;
    o[6] = (_Float16)b.z; o[7] = (_Float16)b.w;
    *(v8h*)(dst + i) = o;
  }
}

// kernel 0b: bias_sum = b_ih + b_hh (512 values, f32)
__global__ void bias_sum_kernel(const float* __restrict__ bi,
                                const float* __restrict__ bh,
                                float* __restrict__ bs) {
  int i = blockIdx.x * blockDim.x + threadIdx.x;
  if (i < G4) bs[i] = bi[i] + bh[i];
}

// ---------------------------------------------------------------------------
// kernel 1: gx[t][n][b] = sum_i x[b][t][i] * W_ih[n][i] + bias_sum[n]  (f16 out)
//
// A = x rows (M=batch, K=I), B = W_ih^T (K=I, N=gate cols), f32 accum.
// grid = (16 batch tiles, 32 col tiles, 8 time slices), 256 thr = 8 waves.
// All 8 A-fragments are preloaded (one 16-load clause) and the K-chain is
// split over two accumulators so WMMA issue overlaps the global loads.
// ---------------------------------------------------------------------------
__global__ void gx_gemm_kernel(const _Float16* __restrict__ xh,
                               const _Float16* __restrict__ wih,
                               const float*    __restrict__ bsum,
                               _Float16* __restrict__ gx) {
  const int bt   = blockIdx.x;          // batch tile 0..15
  const int nt   = blockIdx.y;          // gate-col tile 0..31
  const int ts   = blockIdx.z;          // time slice 0..7 (64 t each)
  const int w    = threadIdx.x >> 5;    // wave 0..7
  const int lane = threadIdx.x & 31;
  const int half = lane >> 4;
  const int l16  = lane & 15;

  const int   n    = nt * 16 + l16;     // lane's output column (B/C layout)
  const float bias = bsum[n];

  // B fragments: lane holds W_ih[n][k0 + half*16 .. +15] -> contiguous 32B
  v16h Bf[8];
#pragma unroll
  for (int c = 0; c < 8; ++c)
    Bf[c] = *(const v16h*)(wih + (long)n * INP + c * 32 + half * 16);

  const int brow = bt * 16 + l16;       // A-matrix row (batch index)

#pragma unroll 1
  for (int it = 0; it < 8; ++it) {
    const int t = ts * 64 + it * 8 + w;
    const _Float16* abase = xh + ((long)brow * TT + t) * INP + half * 16;

    // preload ALL 8 A fragments -> 16 b128 loads pipeline as one clause
    v16h a[8];
#pragma unroll
    for (int c = 0; c < 8; ++c) {
      v8h lo = *(const v8h*)(abase + c * 32);
      v8h hi = *(const v8h*)(abase + c * 32 + 8);
#pragma unroll
      for (int e = 0; e < 8; ++e) { a[c][e] = lo[e]; a[c][e + 8] = hi[e]; }
    }

    // two independent accumulator chains (even / odd K-chunks)
    v8f acc0 = {}, acc1 = {};
#pragma unroll
    for (int c = 0; c < 4; ++c) {
      acc0 = __builtin_amdgcn_wmma_f32_16x16x32_f16(false, a[2 * c],     false,
                                                    Bf[2 * c],     (short)0, acc0,
                                                    false, false);
      acc1 = __builtin_amdgcn_wmma_f32_16x16x32_f16(false, a[2 * c + 1], false,
                                                    Bf[2 * c + 1], (short)0, acc1,
                                                    false, false);
    }

    // C layout: lane col = n, rows = half*8 + v  -> 8 consecutive b, 1 b128 store
    v8h out;
#pragma unroll
    for (int v = 0; v < 8; ++v)
      out[v] = (_Float16)(acc0[v] + acc1[v] + bias);
    *(v8h*)(gx + ((long)t * G4 + n) * BB + bt * 16 + half * 8) = out;
  }
}

// ---------------------------------------------------------------------------
// kernel 2: persistent LSTM scan. One workgroup per batch tile of 16 rows
// (rows of the recurrence are independent -> no inter-WG sync needed).
// W_hh (f16, 128 KB) is DMA'd into LDS by the Tensor Data Mover (one wave
// issues tensor_load_to_lds with a 2-D D#, then s_wait_tensorcnt 0).
// Each wave keeps its 16 W_hh^T B-fragments + c-state in registers for all
// 512 steps; wave w owns hidden cols [16w,16w+16), whose 4 gate tiles are
// exactly the i/f/g/o values its c/h update needs (wave-local elementwise).
// ---------------------------------------------------------------------------
__global__ void lstm_scan_kernel(const _Float16* __restrict__ gx,
                                 const _Float16* __restrict__ whh,
                                 float* __restrict__ hfin) {
  extern __shared__ _Float16 smem[] __attribute__((aligned(32)));
  _Float16* Wl = smem;                // [512][128] f16 = 128 KB @ LDS offset 0
  _Float16* hl = smem + G4 * HID;     // [16][128]  f16 = 4 KB  @ LDS offset 131072

  const int bt   = blockIdx.x;        // batch tile 0..15
  const int tid  = threadIdx.x;
  const int w    = tid >> 5;          // wave 0..7
  const int lane = tid & 31;
  const int half = lane >> 4;
  const int l16  = lane & 15;
  const int j    = w * 16 + l16;      // hidden column owned by this lane

#if USE_TDM
  // --- TDM: one wave DMAs W_hh[512][128] (f16) into LDS offset 0 ------------
  if (tid < 32) {
    const unsigned long long ga = (unsigned long long)whh;
    // D# group0: count=1 | lds_addr=0 | global_addr | type=2 (bits 127:126)
    v4u g0;
    g0[0] = 1u;                                   // count=1, flags=0
    g0[1] = 0u;                                   // lds_addr = 0
    g0[2] = (unsigned int)(ga & 0xffffffffu);     // global_addr[31:0]
    g0[3] = (unsigned int)((ga >> 32) & 0x01ffffffu) | 0x80000000u; // [56:32]|type=2
    // D# group1: data_size=2B; tensor 128 x 512, tile 128 x 512, stride0=128
    v8i g1;
    g1[0] = 0x00010000;        // workgroup_mask=0, data_size=1 (2 bytes)
    g1[1] = (HID << 16);       // tensor_dim0[15:0] in bits [63:48]
    g1[2] = (G4 << 16);        // tensor_dim0[31:16]=0, tensor_dim1[15:0]=512
    g1[3] = (HID << 16);       // tensor_dim1[31:16]=0, tile_dim0=128
    g1[4] = G4;                // tile_dim1=512, tile_dim2=0
    g1[5] = HID;               // tensor_dim0_stride[31:0]=128
    g1[6] = 0;                 // stride0[47:32]=0, tensor_dim1_stride[15:0]=0
    g1[7] = 0;
    v4i z4 = {0, 0, 0, 0};     // groups 2/3 unused (2-D tensor)
    v8i z8 = {0, 0, 0, 0, 0, 0, 0, 0};
    __builtin_amdgcn_tensor_load_to_lds(g0, g1, z4, z4, z8, 0);
  }
#else
  for (int i = tid * 8; i < G4 * HID; i += 256 * 8)
    *(v8h*)(Wl + i) = *(const v8h*)(whh + i);
#endif

  // h0 = 0 (all threads, overlaps with the DMA)
  for (int i = tid * 8; i < 16 * HID; i += 256 * 8) {
    v8h z = {};
    *(v8h*)(hl + i) = z;
  }

#if USE_TDM
  if (tid < 32) __builtin_amdgcn_s_wait_tensorcnt(0);
#endif
  __syncthreads();

  // Preload W_hh^T fragments: 4 gate tiles x 4 K-chunks (128 VGPRs)
  v16h Bf[4][4];
#pragma unroll
  for (int g = 0; g < 4; ++g) {
    const int nabs = g * HID + j;
#pragma unroll
    for (int c = 0; c < 4; ++c)
      Bf[g][c] = *(const v16h*)(Wl + (long)nabs * HID + c * 32 + half * 16);
  }

  float cst[8], hv[8];
#pragma unroll
  for (int v = 0; v < 8; ++v) { cst[v] = 0.0f; hv[v] = 0.0f; }

  const long gx_boff = (long)bt * 16 + half * 8;

#pragma unroll 1
  for (int t = 0; t < TT; ++t) {
    // A fragments: h tile from LDS (rows = local batch = l16)
    v16h a[4];
#pragma unroll
    for (int c = 0; c < 4; ++c) {
      const _Float16* ap = hl + l16 * HID + c * 32 + half * 16;
      v8h lo = *(const v8h*)(ap);
      v8h hi = *(const v8h*)(ap + 8);
#pragma unroll
      for (int e = 0; e < 8; ++e) { a[c][e] = lo[e]; a[c][e + 8] = hi[e]; }
    }

    // C init from gx: one b128 load per gate tile; prefetch next timestep
    v8f acc[4];
#pragma unroll
    for (int g = 0; g < 4; ++g) {
      const _Float16* gp = gx + ((long)t * G4 + g * HID + j) * BB + gx_boff;
      v8h gc = *(const v8h*)gp;
      if (t + 1 < TT) __builtin_prefetch(gp + (long)G4 * BB, 0, 3);
#pragma unroll
      for (int v = 0; v < 8; ++v) acc[g][v] = (float)gc[v];
    }
    __syncthreads();  // all waves done reading h before it is overwritten

    // gates += h @ W_hh^T   (16 WMMA per wave per step, 4 independent chains)
#pragma unroll
    for (int g = 0; g < 4; ++g)
#pragma unroll
      for (int c = 0; c < 4; ++c)
        acc[g] = __builtin_amdgcn_wmma_f32_16x16x32_f16(
            false, a[c], false, Bf[g][c], (short)0, acc[g], false, false);

    // gate math + state update (wave-local; c stays in registers)
#pragma unroll
    for (int v = 0; v < 8; ++v) {
      float ig = sigmoid_f(acc[0][v]);
      float fg = sigmoid_f(acc[1][v]);
      float gg = tanh_f(acc[2][v]);
      float og = sigmoid_f(acc[3][v]);
      float c2 = fg * cst[v] + ig * gg;
      cst[v]   = c2;
      hv[v]    = og * tanh_f(c2);
      hl[(half * 8 + v) * HID + j] = (_Float16)hv[v];
    }
    __syncthreads();  // new h visible to all waves
  }

  // final hidden state (f32) for the MLP head
#pragma unroll
  for (int v = 0; v < 8; ++v) {
    const int babs = bt * 16 + half * 8 + v;
    hfin[(long)babs * HID + j] = hv[v];
  }
}

// ---------------------------------------------------------------------------
// kernel 3: tiny MLP head, one thread per batch row (2.7 MFLOP total)
// ---------------------------------------------------------------------------
__global__ void head_kernel(const float* __restrict__ hfin,
                            const float* __restrict__ W1, const float* __restrict__ b1,
                            const float* __restrict__ W2, const float* __restrict__ b2,
                            const float* __restrict__ W3, const float* __restrict__ b3,
                            float* __restrict__ out) {
  const int b = blockIdx.x * blockDim.x + threadIdx.x;
  if (b >= BB) return;
  float h0[128];
#pragma unroll
  for (int k = 0; k < 128; ++k) h0[k] = hfin[(long)b * 128 + k];
  float h1[64];
  for (int o = 0; o < 64; ++o) {
    float s = b1[o];
    for (int k = 0; k < 128; ++k) s += W1[o * 128 + k] * h0[k];
    h1[o] = fmaxf(s, 0.0f);
  }
  float h2[32];
  for (int o = 0; o < 32; ++o) {
    float s = b2[o];
    for (int k = 0; k < 64; ++k) s += W2[o * 64 + k] * h1[k];
    h2[o] = fmaxf(s, 0.0f);
  }
  for (int o = 0; o < 4; ++o) {
    float s = b3[o];
    for (int k = 0; k < 32; ++k) s += W3[o * 32 + k] * h2[k];
    out[b * 4 + o] = s;
  }
}

// ---------------------------------------------------------------------------
// launch
// ---------------------------------------------------------------------------
extern "C" void kernel_launch(void* const* d_in, const int* in_sizes, int n_in,
                              void* d_out, int out_size, void* d_ws, size_t ws_size,
                              hipStream_t stream) {
  const float* x   = (const float*)d_in[0];
  const float* Wih = (const float*)d_in[1];
  const float* Whh = (const float*)d_in[2];
  const float* bih = (const float*)d_in[3];
  const float* bhh = (const float*)d_in[4];
  const float* W1  = (const float*)d_in[5];
  const float* b1  = (const float*)d_in[6];
  const float* W2  = (const float*)d_in[7];
  const float* b2  = (const float*)d_in[8];
  const float* W3  = (const float*)d_in[9];
  const float* b3  = (const float*)d_in[10];
  float* out = (float*)d_out;

  // workspace layout (~193 MB)
  char* ws = (char*)d_ws;
  _Float16* xh   = (_Float16*)(ws);                              // 64 MiB
  _Float16* gx   = (_Float16*)(ws + (size_t)67108864);           // 128 MiB
  _Float16* wihh = (_Float16*)(ws + (size_t)201326592);          // 256 KiB
  _Float16* whhh = (_Float16*)(ws + (size_t)201588736);          // 128 KiB
  float*    bsum = (float*)   (ws + (size_t)201719808);          // 2 KiB
  float*    hfin = (float*)   (ws + (size_t)201721856);          // 128 KiB

  // 0) convert x, W_ih, W_hh to f16; fold the two biases
  const int nx = BB * TT * INP;                  // 33554432
  cvt_f32_to_f16<<<nx / 8 / 256, 256, 0, stream>>>(x, xh, nx);
  cvt_f32_to_f16<<<(G4 * INP) / 8 / 256, 256, 0, stream>>>(Wih, wihh, G4 * INP);
  cvt_f32_to_f16<<<(G4 * HID) / 8 / 256, 256, 0, stream>>>(Whh, whhh, G4 * HID);
  bias_sum_kernel<<<2, 256, 0, stream>>>(bih, bhh, bsum);

  // 1) input-projection GEMM -> gx[t][4H][B]
  gx_gemm_kernel<<<dim3(16, 32, 8), 256, 0, stream>>>(xh, wihh, bsum, gx);

  // 2) persistent recurrent scan (16 independent workgroups, 132 KB LDS each)
  const int lds_bytes = (G4 * HID + 16 * HID) * (int)sizeof(_Float16); // 135168
  (void)hipFuncSetAttribute(reinterpret_cast<const void*>(lstm_scan_kernel),
                            hipFuncAttributeMaxDynamicSharedMemorySize, lds_bytes);
  lstm_scan_kernel<<<dim3(16), 256, lds_bytes, stream>>>(gx, whhh, hfin);

  // 3) MLP head
  head_kernel<<<1, 256, 0, stream>>>(hfin, W1, b1, W2, b2, W3, b3, out);
}
